// TransformerDiffusion_45105746542792
// MI455X (gfx1250) — matrix operand order repaired and checked by reading, not compiled, not run on hardware
//
#include <hip/hip_runtime.h>
#include <cstdint>

// ---------------------------------------------------------------------------
// CDNA5 (gfx1250) types
// ---------------------------------------------------------------------------
typedef __attribute__((ext_vector_type(16))) __bf16     v16bf;
typedef __attribute__((ext_vector_type(8)))  float      v8f;
typedef __attribute__((ext_vector_type(4)))  unsigned   u32x4;
typedef __attribute__((ext_vector_type(8)))  int        i32x8;
typedef __attribute__((ext_vector_type(4)))  int        i32x4;

#define TI_M 128           // block rows  (all conv out-channels are multiples of 128)
#define TI_N 64            // block cols
#define TI_K 32            // K step (one bf16 WMMA depth)

__device__ __forceinline__ unsigned short f2bf(float f) {
  unsigned u = __builtin_bit_cast(unsigned, f);
  u += 0x7FFFu + ((u >> 16) & 1u);            // round-to-nearest-even
  return (unsigned short)(u >> 16);
}
__device__ __forceinline__ __bf16 bfbits(unsigned short s) {
  return __builtin_bit_cast(__bf16, s);
}
__device__ __forceinline__ float siluf(float x) { return x / (1.f + __expf(-x)); }
__device__ __forceinline__ float geluf(float x) { return 0.5f * x * (1.f + erff(x * 0.7071067811865476f)); }
__device__ __forceinline__ float actf(float x, int a) {
  if (a == 1) return siluf(x);
  if (a == 2) return geluf(x);
  return x;
}

// ---------------------------------------------------------------------------
// WMMA GEMM:  C[M x Ncols] = W[M x K] (fp32, row-major) @ im2col(X) + bias
// X per batch is [Cin x Nin] row-major; GEMM column n pulls
// X[ki/kw, n*stride + ki%kw - pad] (zero padded): kw=1 -> 1x1 conv,
// kw=3/5 -> temporal convs. Weight tiles (128x32 fp32) arrive via the
// Tensor Data Mover (TENSORcnt); activations are staged as bf16 by the block.
// 8 waves; each wave owns a 32x32 output tile = 2 A-frags x 2 B-frags ->
// four v_wmma_f32_16x16x32_bf16 per K step (A/B in contiguous k order, C/D per ISA 7.12.2).
// ---------------------------------------------------------------------------
__global__ __launch_bounds__(256) void k_gemm(
    const float* __restrict__ W, const float* __restrict__ X,
    const float* __restrict__ bias, const float* __restrict__ addsrc,
    float* __restrict__ C,
    int M, int Ncols, int K, int Nin, int kw, int cstride, int pad,
    long long sXb, long long sCb, long long sAddb, int act)
{
  __shared__ float sA[TI_M * TI_K];                 // weight tile (TDM target) 16KB
  __shared__ unsigned short sB[TI_K][TI_N + 8];     // activation tile, bf16
  if ((int)threadIdx.x == -1) sA[0] = 0.f;          // keep TDM-written LDS alive

  const int b = blockIdx.z;
  const float* Xb = X + (long long)b * sXb;
  float*       Cb = C + (long long)b * sCb;
  const float* Ab = addsrc ? addsrc + (long long)b * sAddb : nullptr;

  const int wave = threadIdx.x >> 5;
  const int lane = threadIdx.x & 31;
  const int wm = wave & 3;                 // 4 row-waves  (32 rows each)
  const int wn = wave >> 2;                // 2 col-waves  (32 cols each)
  const int rowTile = blockIdx.y * TI_M;
  const int colTile = blockIdx.x * TI_N;

  v8f acc00 = {}, acc01 = {};              // A0 x {B0,B1}
  v8f acc10 = {}, acc11 = {};              // A1 x {B0,B1}

  const unsigned ldsA = (unsigned)(unsigned long long)(void*)&sA[0];

  for (int k0 = 0; k0 < K; k0 += TI_K) {
    __syncthreads();

    // ---- A tile: Tensor Data Mover, 2-D tile 32 (K) x 128 (M) of fp32 ----
    if (wave == 0) {
      unsigned long long gaddr =
          (unsigned long long)(const void*)(W + (long long)rowTile * K + k0);
      unsigned td0 = (unsigned)(K - k0);          // remaining cols (OOB -> 0)
      unsigned td1 = (unsigned)(M - rowTile);     // remaining rows
      unsigned long long str0 = (unsigned long long)(unsigned)K;
      u32x4 g0;
      g0[0] = 1u;                                              // count=1
      g0[1] = ldsA;                                            // lds_addr
      g0[2] = (unsigned)(gaddr & 0xFFFFFFFFull);               // global_addr lo
      g0[3] = (unsigned)((gaddr >> 32) & 0x1FFFFFFull) | (2u << 30); // hi | type=2
      i32x8 g1;
      g1[0] = (int)(2u << 16);                                 // data_size = 4B
      g1[1] = (int)((td0 & 0xFFFFu) << 16);                    // tensor_dim0 lo
      g1[2] = (int)((td0 >> 16) | ((td1 & 0xFFFFu) << 16));    // dim0 hi | dim1 lo
      g1[3] = (int)((td1 >> 16) | ((unsigned)TI_K << 16));     // dim1 hi | tile_dim0
      g1[4] = (int)((unsigned)TI_M);                           // tile_dim1 (tile_dim2=0)
      g1[5] = (int)(str0 & 0xFFFFFFFFull);                     // dim0_stride lo
      g1[6] = (int)((str0 >> 32) & 0xFFFFull);                 // dim0_stride hi
      g1[7] = 0;
      i32x4 gz = {0, 0, 0, 0};
#if __has_include(<hip/amd_detail/amd_gfx1250_TDM.h>)
      i32x8 gz8 = {0, 0, 0, 0, 0, 0, 0, 0};
      __builtin_amdgcn_tensor_load_to_lds(g0, g1, gz, gz, gz8, 0);
#else
      __builtin_amdgcn_tensor_load_to_lds(g0, g1, gz, gz, 0);
#endif
    }

    // ---- B tile: cooperative im2col gather, fp32 -> bf16 ----
    for (int idx = threadIdx.x; idx < TI_K * TI_N; idx += 256) {
      int kk = idx / TI_N, nn = idx - kk * TI_N;
      int ki = k0 + kk;
      float v = 0.f;
      int pos = colTile + nn;
      if (ki < K && pos < Ncols) {
        int ic = ki / kw, t = ki - ic * kw;
        int sc = pos * cstride + t - pad;
        if (sc >= 0 && sc < Nin) v = Xb[(long long)ic * Nin + sc];
      }
      sB[kk][nn] = f2bf(v);
    }

    if (wave == 0) __builtin_amdgcn_s_wait_tensorcnt(0);
    __syncthreads();

    if (k0 + TI_K < K)                                // pre-touch next weight tile
      __builtin_prefetch(W + (long long)rowTile * K + k0 + TI_K, 0, 1);

    // ---- build fragments (contiguous k: element j of lane half h holds K 16h+j) ----
    v16bf afr0, afr1, bfr0, bfr1;
    {
      int r0 = (lane & 15) + 32 * wm;                 // A frag 0: rows 32*wm..+15
      int kb = (lane >> 4) << 4;                      // lanes 16-31: K 16..31
      #pragma unroll
      for (int j = 0; j < 16; ++j) {
        int kk = j + kb;                              // K 0..15 (+16)
        afr0[j] = bfbits(f2bf(sA[r0 * TI_K + kk]));
        afr1[j] = bfbits(f2bf(sA[(r0 + 16) * TI_K + kk]));
      }
      int c0 = (lane & 15) + 32 * wn;
      int ko = (lane >> 4) << 4;                      // lanes 16-31: K 16..31
      #pragma unroll
      for (int j = 0; j < 16; ++j) {
        bfr0[j] = bfbits(sB[ko + j][c0]);
        bfr1[j] = bfbits(sB[ko + j][c0 + 16]);
      }
    }
    acc00 = __builtin_amdgcn_wmma_f32_16x16x32_bf16(false, afr0, false, bfr0,
                                                    (short)0, acc00, false, false);
    acc01 = __builtin_amdgcn_wmma_f32_16x16x32_bf16(false, afr0, false, bfr1,
                                                    (short)0, acc01, false, false);
    acc10 = __builtin_amdgcn_wmma_f32_16x16x32_bf16(false, afr1, false, bfr0,
                                                    (short)0, acc10, false, false);
    acc11 = __builtin_amdgcn_wmma_f32_16x16x32_bf16(false, afr1, false, bfr1,
                                                    (short)0, acc11, false, false);
  }

  // ---- epilogue: bias + residual + activation ----
  int colBase = colTile + 32 * wn + (lane & 15);
  int rowHalf = (lane >> 4) * 8;                      // C/D: lanes 16-31 -> M+8
  #pragma unroll
  for (int r = 0; r < 2; ++r) {
    #pragma unroll
    for (int i = 0; i < 8; ++i) {
      int row = rowTile + 32 * wm + 16 * r + rowHalf + i;
      if (row >= M) continue;
      float bv = bias ? bias[row] : 0.f;
      float a0 = (r == 0) ? acc00[i] : acc10[i];
      float a1 = (r == 0) ? acc01[i] : acc11[i];
      int c0 = colBase;
      if (c0 < Ncols) {
        float v = a0 + bv;
        if (Ab) v += Ab[(long long)row * Ncols + c0];
        Cb[(long long)row * Ncols + c0] = actf(v, act);
      }
      int c1 = colBase + 16;
      if (c1 < Ncols) {
        float v = a1 + bv;
        if (Ab) v += Ab[(long long)row * Ncols + c1];
        Cb[(long long)row * Ncols + c1] = actf(v, act);
      }
    }
  }
}

// ---------------------------------------------------------------------------
// Attention (ch = 64 in both the main trunk and the cond encoder).
// One thread per query position, streaming online softmax with optional
// local window + learned relative bias pos_emb[window - |t-s|].
// ---------------------------------------------------------------------------
__global__ __launch_bounds__(128) void k_attn(
    const float* __restrict__ qkv, float* __restrict__ out,
    const float* __restrict__ pos_emb, int n, int heads, int window,
    long long sQb, long long sOb)
{
  const int CH = 64;
  int bh = blockIdx.y;
  int b = bh / heads, hh = bh - b * heads;
  int t = blockIdx.x * blockDim.x + threadIdx.x;
  if (t >= n) return;
  const float* base = qkv + (long long)b * sQb + (long long)hh * 3 * CH * n;
  const float* Q  = base;
  const float* Kp = base + (long long)CH * n;
  const float* Vp = base + (long long)2 * CH * n;
  float* ob = out + (long long)b * sOb + (long long)hh * CH * n;

  float q[CH];
  #pragma unroll
  for (int c = 0; c < CH; ++c) q[c] = Q[(long long)c * n + t];

  const float sc = 0.125f;                       // (q*ch^-.25)(k*ch^-.25) = d/8
  float m = -3.0e38f, l = 0.f;
  float o[CH];
  #pragma unroll
  for (int c = 0; c < CH; ++c) o[c] = 0.f;

  int s0 = 0, s1 = n;
  if (window > 0) {
    s0 = t - (window - 1); if (s0 < 0) s0 = 0;
    s1 = t + window;       if (s1 > n) s1 = n;
  }
  for (int s = s0; s < s1; ++s) {
    float d = 0.f;
    #pragma unroll
    for (int c = 0; c < CH; ++c) d += q[c] * Kp[(long long)c * n + s];
    float a = d * sc;
    if (pos_emb) {
      int dd = t - s; if (dd < 0) dd = -dd;
      a += pos_emb[window - dd];
    }
    float nm = fmaxf(m, a);
    float f = __expf(m - nm);
    float e = __expf(a - nm);
    l = l * f + e;
    #pragma unroll
    for (int c = 0; c < CH; ++c) o[c] = o[c] * f + e * Vp[(long long)c * n + s];
    m = nm;
  }
  float inv = 1.f / l;
  #pragma unroll
  for (int c = 0; c < CH; ++c) ob[(long long)c * n + t] = o[c] * inv;
}

// ---------------------------------------------------------------------------
// GroupNorm (+ optional SiLU).  grid = (groups, B)
// ---------------------------------------------------------------------------
__global__ __launch_bounds__(256) void k_groupnorm(
    const float* __restrict__ x, float* __restrict__ y,
    const float* __restrict__ gamma, const float* __restrict__ beta,
    int C, int n, int groups, int act, long long sXb, long long sYb)
{
  int g = blockIdx.x, b = blockIdx.y;
  int cpg = C / groups;
  const float* xb = x + (long long)b * sXb + (long long)g * cpg * n;
  float*       yb = y + (long long)b * sYb + (long long)g * cpg * n;
  long long cnt = (long long)cpg * n;
  float s = 0.f, s2 = 0.f;
  for (long long i = threadIdx.x; i < cnt; i += 256) {
    float v = xb[i]; s += v; s2 += v * v;
  }
  __shared__ float rs[256], rq[256];
  rs[threadIdx.x] = s; rq[threadIdx.x] = s2;
  __syncthreads();
  for (int st = 128; st > 0; st >>= 1) {
    if ((int)threadIdx.x < st) {
      rs[threadIdx.x] += rs[threadIdx.x + st];
      rq[threadIdx.x] += rq[threadIdx.x + st];
    }
    __syncthreads();
  }
  float mu  = rs[0] / (float)cnt;
  float var = rq[0] / (float)cnt - mu * mu;
  float inv = rsqrtf(var + 1e-5f);
  for (long long i = threadIdx.x; i < cnt; i += 256) {
    int c = g * cpg + (int)(i / n);
    float v = (xb[i] - mu) * inv * gamma[c] + beta[c];
    yb[i] = actf(v, act);
  }
}

// ---------------------------------------------------------------------------
// small elementwise helpers
// ---------------------------------------------------------------------------
__global__ void k_cglu(const float* __restrict__ h2, float* __restrict__ out,
                       int cmid, int n, long long sHb, long long sOb)
{
  int b = blockIdx.y;
  long long total = (long long)cmid * n;
  for (long long i = blockIdx.x * (long long)blockDim.x + threadIdx.x; i < total;
       i += (long long)gridDim.x * blockDim.x) {
    float a = h2[(long long)b * sHb + i];
    float g = h2[(long long)b * sHb + total + i];
    out[(long long)b * sOb + i] = a * geluf(g);
  }
}

__global__ void k_copy(const float* __restrict__ src, float* __restrict__ dst,
                       long long count, long long sSb, long long sDb)
{
  int b = blockIdx.y;
  for (long long i = blockIdx.x * (long long)blockDim.x + threadIdx.x; i < count;
       i += (long long)gridDim.x * blockDim.x)
    dst[(long long)b * sDb + i] = src[(long long)b * sSb + i];
}

__global__ void k_bcast(const float* __restrict__ emb, float* __restrict__ dst,
                        int rows, int n, int embStride, long long sDb)
{
  int b = blockIdx.y;
  long long total = (long long)rows * n;
  for (long long i = blockIdx.x * (long long)blockDim.x + threadIdx.x; i < total;
       i += (long long)gridDim.x * blockDim.x) {
    int r = (int)(i / n);
    dst[(long long)b * sDb + i] = emb[b * embStride + r];
  }
}

__global__ void k_interp(const float* __restrict__ src, float* __restrict__ dst,
                         int C, int m, int nout)
{
  int b = blockIdx.y;
  long long total = (long long)C * nout;
  for (long long idx = blockIdx.x * (long long)blockDim.x + threadIdx.x; idx < total;
       idx += (long long)gridDim.x * blockDim.x) {
    int c = (int)(idx / nout), i = (int)(idx % nout);
    float pos = (float)i * (float)(m - 1) / (float)(nout - 1);
    int lo = (int)floorf(pos);
    int hi = lo + 1; if (hi > m - 1) hi = m - 1;
    float fr = pos - (float)lo;
    const float* sb = src + (long long)b * C * m + (long long)c * m;
    dst[(long long)b * C * nout + idx] = sb[lo] * (1.f - fr) + sb[hi] * fr;
  }
}

// timestep_embedding (dim 256) -> silu-MLP (256) -> 64 outputs into blk_emb
__global__ __launch_bounds__(256) void k_time_mlp(
    const int* __restrict__ tsteps,
    const float* __restrict__ w1, const float* __restrict__ b1,
    const float* __restrict__ w2, const float* __restrict__ b2,
    float* __restrict__ blk, int blkStride, int off)
{
  int b = blockIdx.x;
  __shared__ float emb[256], h1[256];
  float t = (float)tsteps[b];
  for (int i = threadIdx.x; i < 128; i += blockDim.x) {
    float f = __expf(-logf(10000.f) * (float)i / 128.f);
    emb[i]       = cosf(t * f);
    emb[i + 128] = sinf(t * f);
  }
  __syncthreads();
  for (int j = threadIdx.x; j < 256; j += blockDim.x) {
    float acc = b1[j];
    for (int i = 0; i < 256; ++i) acc += emb[i] * w1[j * 256 + i];
    h1[j] = siluf(acc);
  }
  __syncthreads();
  for (int o = threadIdx.x; o < 64; o += blockDim.x) {
    float acc = b2[o];
    for (int j = 0; j < 256; ++j) acc += h1[j] * w2[o * 256 + j];
    blk[b * blkStride + off + o] = acc;
  }
}

__global__ void k_gather_res(const float* __restrict__ tbl, const int* __restrict__ res,
                             float* __restrict__ dst, int dim, int dstStride, int off)
{
  int b = blockIdx.x;
  for (int i = threadIdx.x; i < dim; i += blockDim.x)
    dst[b * dstStride + off + i] = tbl[res[b] * dim + i];
}

__global__ void k_addres(const float* __restrict__ tbl, const int* __restrict__ res,
                         float* __restrict__ h, int C, int n, long long sHb)
{
  int b = blockIdx.y;
  long long total = (long long)C * n;
  for (long long i = blockIdx.x * (long long)blockDim.x + threadIdx.x; i < total;
       i += (long long)gridDim.x * blockDim.x) {
    int c = (int)(i / n);
    h[(long long)b * sHb + i] += tbl[res[b] * C + c];
  }
}

// code_emb[b, o] = sum_c h[b, c, 0] * W[o, c]
__global__ __launch_bounds__(256) void k_cond_out(
    const float* __restrict__ h, const float* __restrict__ W,
    float* __restrict__ blk, int Cin, int n, long long sHb,
    int outDim, int blkStride, int off)
{
  int b = blockIdx.x;
  for (int o = threadIdx.x; o < outDim; o += blockDim.x) {
    float acc = 0.f;
    for (int c = 0; c < Cin; ++c)
      acc += h[(long long)b * sHb + (long long)c * n] * W[o * Cin + c];
    blk[b * blkStride + off + o] = acc;
  }
}

__global__ void k_zero(float* __restrict__ p, long long nn)
{
  for (long long i = blockIdx.x * (long long)blockDim.x + threadIdx.x; i < nn;
       i += (long long)gridDim.x * blockDim.x)
    p[i] = 0.f;
}

// ---------------------------------------------------------------------------
// host-side parameter bundles
// ---------------------------------------------------------------------------
struct FFp  { const float *w, *b, *gg, *gb, *gw, *gbias; };
struct Sub  { const float *ng, *nb, *qw, *qb, *pw, *pb, *xw, *xb, *pos; FFp f1, f2; };
struct Layer{ const float *pg, *pb; Sub b1, b2; const float *ow; };
struct CondBlk {
  const float *ng, *nb, *qw, *qb, *pw, *pb;
  const float *n1g, *n1b, *w1, *b1, *n2g, *n2b, *w2, *b2;
};

extern "C" void kernel_launch(void* const* d_in, const int* in_sizes, int n_in,
                              void* d_out, int out_size, void* d_ws, size_t ws_size,
                              hipStream_t stream)
{
  (void)in_sizes; (void)n_in; (void)out_size;
  const int Bb = 8, Nn = 384, NC = 198;   // NC = (400 - 5)/2 + 1

  const float* x        = (const float*)d_in[0];   // [8,1024,384]
  const float* x_prior  = (const float*)d_in[1];   // [8,256,96]
  const float* cond_in  = (const float*)d_in[2];   // [8,256,400]
  const int*   tsteps   = (const int*)d_in[3];
  const int*   ptsteps  = (const int*)d_in[4];
  const int*   resol    = (const int*)d_in[5];

  // ---- walk the params pytree (insertion order) ----
  int ip = 6;
  auto F = [&]() -> const float* { return (const float*)d_in[ip++]; };

  const float *tm_w1 = F(), *tm_b1 = F(), *tm_w2 = F(), *tm_b2 = F();
  const float *pm_w1 = F(), *pm_b1 = F(), *pm_w2 = F(), *pm_b2 = F();
  const float *res_embed = F();
  const float *c_init_w = F(), *c_init_b = F(), *c_res_emb = F();
  CondBlk cb[6];
  for (int i = 0; i < 6; ++i) {
    cb[i].ng = F(); cb[i].nb = F(); cb[i].qw = F(); cb[i].qb = F();
    cb[i].pw = F(); cb[i].pb = F();
    cb[i].n1g = F(); cb[i].n1b = F(); cb[i].w1 = F(); cb[i].b1 = F();
    cb[i].n2g = F(); cb[i].n2b = F(); cb[i].w2 = F(); cb[i].b2 = F();
  }
  const float *c_out_w = F();
  const float *inp_w = F(), *inp_b = F();
  Layer L[8];
  auto loadFF  = [&](FFp& f) { f.w = F(); f.b = F(); f.gg = F(); f.gb = F();
                               f.gw = F(); f.gbias = F(); };
  auto loadSub = [&](Sub& s) { s.ng = F(); s.nb = F(); s.qw = F(); s.qb = F();
                               s.pw = F(); s.pb = F(); s.xw = F(); s.xb = F();
                               s.pos = F(); loadFF(s.f1); loadFF(s.f2); };
  for (int i = 0; i < 8; ++i) {
    L[i].pg = F(); L[i].pb = F();
    loadSub(L[i].b1); loadSub(L[i].b2);
    L[i].ow = F();
  }
  const float *out_g = F(), *out_b = F(), *out_w = F(), *out_b2 = F();

  // ---- workspace layout ----
  float* ws = (float*)d_ws;
  size_t off = 0;
  auto alloc = [&](size_t nf) { float* p = ws + off; off += nf; return p; };
  float* xp    = alloc((size_t)Bb * 256 * Nn);
  float* hin   = alloc((size_t)Bb * 1280 * Nn);
  float* h     = alloc((size_t)Bb * 1024 * Nn);
  float* gbuf  = alloc((size_t)Bb * 2496 * Nn);
  float* cat1  = alloc((size_t)Bb * 2496 * Nn);
  float* cat2  = alloc((size_t)Bb * 2496 * Nn);
  float* qkvb  = alloc((size_t)Bb * 768 * Nn);
  float* attnb = alloc((size_t)Bb * 256 * Nn);
  float* t128  = alloc((size_t)Bb * 128 * Nn);
  float* t256  = alloc((size_t)Bb * 256 * Nn);
  float* blk   = alloc((size_t)Bb * 448);
  float* ch_   = alloc((size_t)Bb * 1024 * NC);
  float* cgn   = alloc((size_t)Bb * 1024 * NC);
  float* ct    = alloc((size_t)Bb * 1024 * NC);
  float* cqkv  = alloc((size_t)Bb * 3072 * NC);

  if (off * sizeof(float) > ws_size) {          // deterministic fallback
    k_zero<<<1024, 256, 0, stream>>>((float*)d_out, (long long)Bb * 512 * Nn);
    return;
  }

  const long long SH = (long long)1024 * Nn;    // h batch stride
  const long long S1 = (long long)2496 * Nn;    // cat/gbuf batch stride
  const long long SC = (long long)1024 * NC;    // cond batch stride

  auto gemm = [&](const float* W, const float* X, const float* bias,
                  const float* add, float* C, int M, int Ncols, int K, int Nin,
                  int kw, int cstride, int pad, long long sXb, long long sCb,
                  long long sAddb, int act) {
    dim3 grid((Ncols + TI_N - 1) / TI_N, M / TI_M, Bb);
    k_gemm<<<grid, 256, 0, stream>>>(W, X, bias, add, C, M, Ncols, K, Nin, kw,
                                     cstride, pad, sXb, sCb, sAddb, act);
  };
  auto gn = [&](const float* X, float* Y, const float* g_, const float* b_,
                int C, int n, int groups, int act, long long sXb, long long sYb) {
    dim3 grid(groups, Bb);
    k_groupnorm<<<grid, 256, 0, stream>>>(X, Y, g_, b_, C, n, groups, act, sXb, sYb);
  };
  auto copyRows = [&](const float* src, float* dst, long long cnt,
                      long long sSb, long long sDb) {
    dim3 grid((unsigned)((cnt + 255) / 256), Bb);
    k_copy<<<grid, 256, 0, stream>>>(src, dst, cnt, sSb, sDb);
  };

  // ---- input path: xp = interp(x_prior), hin = cat(x, xp), h = inp conv ----
  { dim3 grid(96, Bb); k_interp<<<grid, 256, 0, stream>>>(x_prior, xp, 256, 96, Nn); }
  copyRows(x, hin, (long long)1024 * Nn, (long long)1024 * Nn, (long long)1280 * Nn);
  copyRows(xp, hin + (long long)1024 * Nn, (long long)256 * Nn,
           (long long)256 * Nn, (long long)1280 * Nn);
  gemm(inp_w, hin, inp_b, nullptr, h, 1024, Nn, 1280 * 3, Nn, 3, 1, 1,
       (long long)1280 * Nn, SH, 0, 0);

  // ---- blk_emb = [te | pte | res_embed | code_emb]  (448) ----
  k_time_mlp<<<Bb, 256, 0, stream>>>(tsteps, tm_w1, tm_b1, tm_w2, tm_b2, blk, 448, 0);
  k_time_mlp<<<Bb, 256, 0, stream>>>(ptsteps, pm_w1, pm_b1, pm_w2, pm_b2, blk, 448, 64);
  k_gather_res<<<Bb, 64, 0, stream>>>(res_embed, resol, blk, 64, 448, 128);

  // ---- conditioning encoder ----
  gemm(c_init_w, cond_in, c_init_b, nullptr, ch_, 1024, NC, 256 * 5, 400, 5, 2, 0,
       (long long)256 * 400, SC, 0, 0);
  { dim3 grid(792, Bb);
    k_addres<<<grid, 256, 0, stream>>>(c_res_emb, resol, ch_, 1024, NC, SC); }
  for (int i = 0; i < 6; ++i) {
    gn(ch_, cgn, cb[i].ng, cb[i].nb, 1024, NC, 32, 0, SC, SC);
    gemm(cb[i].qw, cgn, cb[i].qb, nullptr, cqkv, 3072, NC, 1024, NC, 1, 1, 0,
         SC, (long long)3072 * NC, 0, 0);
    { dim3 grid((NC + 127) / 128, Bb * 16);
      k_attn<<<grid, 128, 0, stream>>>(cqkv, ct, nullptr, NC, 16, 0,
                                       (long long)3072 * NC, SC); }
    gemm(cb[i].pw, ct, cb[i].pb, ch_, ch_, 1024, NC, 1024, NC, 1, 1, 0,
         SC, SC, SC, 0);                                      // x + proj(attn)
    gn(ch_, cgn, cb[i].n1g, cb[i].n1b, 1024, NC, 32, 1, SC, SC);
    gemm(cb[i].w1, cgn, cb[i].b1, nullptr, ct, 1024, NC, 1024, NC, 1, 1, 0,
         SC, SC, 0, 0);
    gn(ct, cgn, cb[i].n2g, cb[i].n2b, 1024, NC, 32, 1, SC, SC);
    gemm(cb[i].w2, cgn, cb[i].b2, ch_, ch_, 1024, NC, 1024 * 3, NC, 3, 1, 1,
         SC, SC, SC, 0);                                      // x + res conv
  }
  k_cond_out<<<Bb, 256, 0, stream>>>(ch_, c_out_w, blk, 1024, NC, SC, 256, 448, 192);

  // ---- sub_block: attn + concat + ff1 + ff2 ----
  auto sub = [&](const Sub& S, const float* Xb, long long sXb, int Ci,
                 float* Ob, long long sOb) {
    gn(Xb, gbuf, S.ng, S.nb, Ci, Nn, 32, 0, sXb, S1);
    gemm(S.qw, gbuf, S.qb, nullptr, qkvb, 768, Nn, Ci, Nn, 1, 1, 0,
         S1, (long long)768 * Nn, 0, 0);
    { dim3 grid((Nn + 127) / 128, Bb * 4);
      k_attn<<<grid, 128, 0, stream>>>(qkvb, attnb, S.pos, Nn, 4, 64,
                                       (long long)768 * Nn, (long long)256 * Nn); }
    gemm(S.xw, Xb, S.xb, nullptr, t256, 256, Nn, Ci, Nn, 1, 1, 0,
         sXb, (long long)256 * Nn, 0, 0);                     // xp projection
    gemm(S.pw, attnb, S.pb, t256, Ob, 256, Nn, 256, Nn, 1, 1, 0,
         (long long)256 * Nn, sOb, (long long)256 * Nn, 0);   // rows [0,256)
    copyRows(Xb, Ob + (long long)256 * Nn, (long long)Ci * Nn, sXb, sOb);
    int C1 = 256 + Ci;
    gemm(S.f1.w, Ob, S.f1.b, nullptr, t128, 128, Nn, C1, Nn, 1, 1, 0,
         sOb, (long long)128 * Nn, 0, 0);
    gn(t128, gbuf, S.f1.gg, S.f1.gb, 128, Nn, 8, 0, (long long)128 * Nn, S1);
    gemm(S.f1.gw, gbuf, S.f1.gbias, nullptr, t256, 256, Nn, 128, Nn, 1, 1, 0,
         S1, (long long)256 * Nn, 0, 0);
    { dim3 grid((unsigned)(((long long)128 * Nn + 255) / 256), Bb);
      k_cglu<<<grid, 256, 0, stream>>>(t256, Ob + (long long)C1 * Nn, 128, Nn,
                                       (long long)256 * Nn, sOb); }
    int C2 = C1 + 128;
    gemm(S.f2.w, Ob, S.f2.b, nullptr, t128, 128, Nn, C2 * 3, Nn, 3, 1, 1,
         sOb, (long long)128 * Nn, 0, 0);
    gn(t128, gbuf, S.f2.gg, S.f2.gb, 128, Nn, 8, 0, (long long)128 * Nn, S1);
    gemm(S.f2.gw, gbuf, S.f2.gbias, nullptr, t256, 256, Nn, 128, Nn, 1, 1, 0,
         S1, (long long)256 * Nn, 0, 0);
    { dim3 grid((unsigned)(((long long)128 * Nn + 255) / 256), Bb);
      k_cglu<<<grid, 256, 0, stream>>>(t256, Ob + (long long)C2 * Nn, 128, Nn,
                                       (long long)256 * Nn, sOb); }
  };

  // ---- 8 concat-attn layers ----
  for (int li = 0; li < 8; ++li) {
    gn(h, cat1, L[li].pg, L[li].pb, 1024, Nn, 8, 0, SH, S1);
    { dim3 grid((unsigned)(((long long)448 * Nn + 255) / 256), Bb);
      k_bcast<<<grid, 256, 0, stream>>>(blk, cat1 + (long long)1024 * Nn,
                                        448, Nn, 448, S1); }
    sub(L[li].b1, cat1, S1, 1472, cat2, S1);                  // 1472 -> 1984
    sub(L[li].b2, cat2, S1, 1984, cat1, S1);                  // 1984 -> 2496
    gemm(L[li].ow, cat1 + (long long)1472 * Nn, nullptr, h, h,
         1024, Nn, 1024, Nn, 1, 1, 0, S1, SH, SH, 0);         // h += out(last 1024)
  }

  // ---- output head ----
  gn(h, gbuf, out_g, out_b, 1024, Nn, 32, 1, SH, S1);         // gn + silu
  gemm(out_w, gbuf, out_b2, nullptr, (float*)d_out, 512, Nn, 1024 * 3, Nn,
       3, 1, 1, S1, (long long)512 * Nn, 0, 0);
}
